// ImprovedVectorQuantizer_72584947303065
// MI455X (gfx1250) — matrix operand order repaired and checked by reading, not compiled, not run on hardware
//
#include <hip/hip_runtime.h>
#include <hip/hip_bf16.h>

typedef __attribute__((ext_vector_type(16))) __bf16 v16bf;
typedef __attribute__((ext_vector_type(8)))  float  v8f;
typedef __attribute__((ext_vector_type(4)))  float  v4f;

#define NUM_EMB 1024
#define DIM     256
#define NTOK    65536          // 64*32*32
#define NELEM   16777216       // NTOK*DIM
#define BROW    272            // LDS row stride in bf16 (256 + 16 pad, keeps 32B align, spreads banks)

// --------------------------------------------------------------------------
// K0: per-code squared norms; zero histogram + loss accumulator.
// --------------------------------------------------------------------------
__global__ void vq_init(const float* __restrict__ embed,
                        float* __restrict__ enorm,
                        int* __restrict__ counts,
                        float* __restrict__ lossAcc) {
    int tid = blockIdx.x * blockDim.x + threadIdx.x;   // 1024 threads total
    if (tid < NUM_EMB) {
        const float* p = embed + tid * DIM;
        float s = 0.f;
        #pragma unroll 4
        for (int i = 0; i < DIM; i += 4) {
            v4f v = *(const v4f*)(p + i);
            s += v.x * v.x + v.y * v.y + v.z * v.z + v.w * v.w;
        }
        enorm[tid]  = s;
        counts[tid] = 0;
    }
    if (tid == 0) *lossAcc = 0.f;
}

// --------------------------------------------------------------------------
// K1: bf16 WMMA distance matmul + running argmin.
// Block = 256 threads = 8 waves; each wave owns 16 tokens (block: 128 tokens).
// All waves iterate the same 64 code tiles; each tile staged once into LDS.
// B slices are preloaded into a register panel so the 8 WMMAs issue
// back-to-back behind a single s_wait_dscnt.
// --------------------------------------------------------------------------
__global__ __launch_bounds__(256) void vq_argmin(
        const float* __restrict__ x,
        const float* __restrict__ embed,
        const float* __restrict__ enorm,
        int*   __restrict__ idx,
        float* __restrict__ outIdxF,
        int*   __restrict__ counts) {
    __shared__ alignas(32) __bf16 Btile[16 * BROW];    // 8.5 KB

    const int lane   = threadIdx.x & 31;
    const int wave   = threadIdx.x >> 5;
    const int rowSel = lane & 15;       // N column within tile / M row selector
    const int hiHalf = lane >> 4;       // 0: lanes 0-15, 1: lanes 16-31
    const int t0     = (blockIdx.x * 8 + wave) * 16;   // first token of this wave

    // ---- Load A panel (16 tokens x 256 dims) as bf16, contiguous-k A layout (same k order as B):
    // lane l (M = l%16): halves 0..15 -> K = 32s + (l/16)*16 + h
    v16bf a[8];
    {
        const float* base = x + (size_t)(t0 + rowSel) * DIM + hiHalf * 16;
        #pragma unroll
        for (int s = 0; s < 8; ++s) {
            const float* p = base + 32 * s;
            v4f v0 = *(const v4f*)(p);
            v4f v1 = *(const v4f*)(p + 4);
            v4f v2 = *(const v4f*)(p + 8);
            v4f v3 = *(const v4f*)(p + 12);
            v16bf av;
            av[0]  = (__bf16)v0.x; av[1]  = (__bf16)v0.y; av[2]  = (__bf16)v0.z; av[3]  = (__bf16)v0.w;
            av[4]  = (__bf16)v1.x; av[5]  = (__bf16)v1.y; av[6]  = (__bf16)v1.z; av[7]  = (__bf16)v1.w;
            av[8]  = (__bf16)v2.x; av[9]  = (__bf16)v2.y; av[10] = (__bf16)v2.z; av[11] = (__bf16)v2.w;
            av[12] = (__bf16)v3.x; av[13] = (__bf16)v3.y; av[14] = (__bf16)v3.z; av[15] = (__bf16)v3.w;
            a[s] = av;
        }
    }

    float minv[8];
    int   mini[8];
    #pragma unroll
    for (int r = 0; r < 8; ++r) { minv[r] = 3.4e38f; mini[r] = 0; }

    for (int nt = 0; nt < 64; ++nt) {
        const int n0 = nt * 16;

        __syncthreads();   // LDS reuse guard
        {   // Cooperative stage: 16 codes x 256 dims f32 -> bf16 LDS tile.
            const int code  = threadIdx.x >> 4;   // 0..15
            const int chunk = threadIdx.x & 15;   // 16 floats each
            const float* p = embed + (size_t)(n0 + code) * DIM + chunk * 16;
            __bf16* q = Btile + code * BROW + chunk * 16;
            #pragma unroll
            for (int i = 0; i < 16; i += 4) {
                v4f v = *(const v4f*)(p + i);
                q[i + 0] = (__bf16)v.x; q[i + 1] = (__bf16)v.y;
                q[i + 2] = (__bf16)v.z; q[i + 3] = (__bf16)v.w;
            }
        }
        __syncthreads();

        // ---- Preload all 8 B slices (bf16 B layout: lane l -> N=l%16,
        // K = 32s + (l/16)*16 + h, h=0..15 -> 32 contiguous bytes per lane),
        // then run the 8 WMMAs back-to-back on the accumulator chain.
        const __bf16* brow = Btile + rowSel * BROW + hiHalf * 16;
        v16bf b[8];
        #pragma unroll
        for (int s = 0; s < 8; ++s)
            b[s] = *(const v16bf*)(brow + 32 * s);

        v8f c = {};
        #pragma unroll
        for (int s = 0; s < 8; ++s)
            c = __builtin_amdgcn_wmma_f32_16x16x32_bf16(
                    false, a[s], false, b[s], (short)0, c, false, false);

        // dist = ||e||^2 - 2 x.e  (||x||^2 is row-constant: irrelevant to argmin)
        const int   col = n0 + rowSel;
        const float en  = enorm[col];
        #pragma unroll
        for (int r = 0; r < 8; ++r) {
            float d = __builtin_fmaf(-2.0f, c[r], en);
            if (d < minv[r]) { minv[r] = d; mini[r] = col; }
        }
    }

    // ---- Reduce over the 16 lanes that hold one row's columns.
    // C layout: VGPR r -> (M=r, lanes 0-15) and (M=r+8, lanes 16-31).
    #pragma unroll
    for (int r = 0; r < 8; ++r) {
        float v = minv[r];
        int   i = mini[r];
        #pragma unroll
        for (int off = 1; off < 16; off <<= 1) {
            float ov = __shfl_xor(v, off, 32);
            int   oi = __shfl_xor(i, off, 32);
            if (ov < v || (ov == v && oi < i)) { v = ov; i = oi; }  // first-occurrence tie-break
        }
        if (rowSel == 0) {
            int row = t0 + r + hiHalf * 8;
            idx[row]     = i;
            outIdxF[row] = (float)i;
            atomicAdd(&counts[i], 1);
        }
    }
}

// --------------------------------------------------------------------------
// K2: gather codebook rows (straight-through output) + commitment loss sum.
// One block per token, one thread per dim.
// --------------------------------------------------------------------------
__global__ __launch_bounds__(256) void vq_gather(
        const float* __restrict__ x,
        const float* __restrict__ embed,
        const int*   __restrict__ idx,
        float* __restrict__ outQ,
        float* __restrict__ lossAcc) {
    __shared__ float red[8];
    const int t = blockIdx.x;
    const int c = threadIdx.x;
    const int code = idx[t];
    const float xi = x[(size_t)t * DIM + c];
    const float q  = embed[(size_t)code * DIM + c];
    outQ[(size_t)t * DIM + c] = xi + (q - xi);   // straight-through numerics
    float d = q - xi;
    float s = d * d;
    #pragma unroll
    for (int off = 16; off; off >>= 1) s += __shfl_xor(s, off, 32);
    if ((threadIdx.x & 31) == 0) red[threadIdx.x >> 5] = s;
    __syncthreads();
    if (threadIdx.x < 8) {
        float v = red[threadIdx.x];
        #pragma unroll
        for (int off = 4; off; off >>= 1) v += __shfl_xor(v, off, 32);
        if (threadIdx.x == 0) atomicAdd(lossAcc, v);
    }
}

// --------------------------------------------------------------------------
// K3: finalize loss + perplexity. Single block, 1024 threads (1 per code).
// --------------------------------------------------------------------------
__global__ __launch_bounds__(1024) void vq_final(
        const int*   __restrict__ counts,
        const float* __restrict__ lossAcc,
        float* __restrict__ outLoss,
        float* __restrict__ outPerp) {
    __shared__ float red[32];
    const int tid = threadIdx.x;
    float p = (float)counts[tid] * (1.0f / (float)NTOK);
    float e = -p * logf(p + 1e-10f);
    #pragma unroll
    for (int off = 16; off; off >>= 1) e += __shfl_xor(e, off, 32);
    if ((tid & 31) == 0) red[tid >> 5] = e;
    __syncthreads();
    if (tid < 32) {
        float v = red[tid];
        #pragma unroll
        for (int off = 16; off; off >>= 1) v += __shfl_xor(v, off, 32);
        if (tid == 0) {
            *outPerp = expf(v);
            *outLoss = 0.25f * (*lossAcc) * (1.0f / (float)NELEM);
        }
    }
}

// --------------------------------------------------------------------------
extern "C" void kernel_launch(void* const* d_in, const int* in_sizes, int n_in,
                              void* d_out, int out_size, void* d_ws, size_t ws_size,
                              hipStream_t stream) {
    const float* x     = (const float*)d_in[0];   // [65536, 256]
    const float* embed = (const float*)d_in[1];   // [1024, 256]

    float* out     = (float*)d_out;
    float* outQ    = out;                 // 16,777,216
    float* outLoss = out + NELEM;         // 1
    float* outPerp = out + NELEM + 1;     // 1
    float* outIdxF = out + NELEM + 2;     // 65,536 (indices as float)

    char*  ws      = (char*)d_ws;
    float* enorm   = (float*)ws;                  // 1024 f32
    int*   counts  = (int*)(ws + 4096);           // 1024 i32
    float* lossAcc = (float*)(ws + 8192);         // 1 f32
    int*   idx     = (int*)(ws + 8448);           // 65536 i32

    vq_init  <<<4, 256, 0, stream>>>(embed, enorm, counts, lossAcc);
    vq_argmin<<<NTOK / 128, 256, 0, stream>>>(x, embed, enorm, idx, outIdxF, counts);
    vq_gather<<<NTOK, 256, 0, stream>>>(x, embed, idx, outQ, lossAcc);
    vq_final <<<1, 1024, 0, stream>>>(counts, lossAcc, outLoss, outPerp);
}
